// XCA_75428215652762
// MI455X (gfx1250) — matrix operand order repaired and checked by reading, not compiled, not run on hardware
//
#include <hip/hip_runtime.h>
#include <hip/hip_bf16.h>

// ---------------------------------------------------------------------------
// XCA forward for MI455X (gfx1250, wave32).
// fp32 math emulated via bf16x3 split on v_wmma_f32_16x16x32_bf16.
// GEMM staging via Tensor Data Mover (tensor_load_to_lds) with hardware
// LDS-row padding, double-buffered against the WMMA pipeline.
// ---------------------------------------------------------------------------

typedef __attribute__((ext_vector_type(16))) __bf16 v16bf;
typedef __attribute__((ext_vector_type(8)))  float  v8f;
typedef unsigned int u32x4 __attribute__((ext_vector_type(4)));
typedef int          i32x4 __attribute__((ext_vector_type(4)));
typedef int          i32x8 __attribute__((ext_vector_type(8)));

#define B_   32
#define N_   1024
#define C_   768
#define H_   16
#define DH_  48
#define K3C  2304   // 3*C

#define LDA  40     // padded LDS row stride in bf16 units (80B, 16B aligned)
#define TILE_USH ((64 + 64 + 128 + 128) * LDA)   // bf16 elems per stage buffer

static __device__ __forceinline__ unsigned short f2bf(float f) {
  unsigned int u = __float_as_uint(f);
  u += 0x7FFFu + ((u >> 16) & 1u);          // round-to-nearest-even
  return (unsigned short)(u >> 16);
}
static __device__ __forceinline__ float bf2f(unsigned short s) {
  return __uint_as_float(((unsigned int)s) << 16);
}

// ---------------------------------------------------------------------------
// Split f32 -> hi/lo bf16 (x = hi + lo to ~2^-16 relative accuracy)
// ---------------------------------------------------------------------------
__global__ __launch_bounds__(256) void split_kernel(
    const float* __restrict__ src,
    unsigned short* __restrict__ hi,
    unsigned short* __restrict__ lo, long n) {
  long i = (long)blockIdx.x * blockDim.x + threadIdx.x;
  long stride = (long)gridDim.x * blockDim.x;
  for (; i < n; i += stride) {
    float v = src[i];
    unsigned short h = f2bf(v);
    hi[i] = h;
    lo[i] = f2bf(v - bf2f(h));
  }
}

#if __has_builtin(__builtin_amdgcn_tensor_load_to_lds)
#define USE_TDM 1
#else
#define USE_TDM 0
#endif

#if USE_TDM
// Issue one TDM 2D tile copy: rows x 32 bf16 elems, global row stride K elems,
// LDS rows padded to 80B by the TDM pad engine (16 DWORDs data + 4 DWORDs pad).
static __device__ __forceinline__ void tdm_tile(
    const unsigned short* gsrc, unsigned ldsByteAddr, int rows, int K) {
  const unsigned long long ga = (unsigned long long)gsrc;
  u32x4 g0;
  g0[0] = 1u;                                           // count=1, user mode
  g0[1] = ldsByteAddr;                                  // lds_addr (bytes)
  g0[2] = (unsigned)(ga & 0xFFFFFFFFu);                 // global_addr[31:0]
  g0[3] = (unsigned)((ga >> 32) & 0x01FFFFFFu) | (2u << 30);  // [56:32] | type=2
  i32x8 g1;
  // data_size=1(2B), pad_enable=1, pad_interval=3(16 DW), pad_amount=3(4 DW)
  g1[0] = (1 << 16) | (1 << 20) | (3 << 22) | (3 << 25);
  g1[1] = 32 << 16;          // tensor_dim0 = 32 (bits[79:48], low half)
  g1[2] = rows << 16;        // tensor_dim1 = rows (bits[111:80], low half)
  g1[3] = 32 << 16;          // tile_dim0 = 32 (bits[127:112])
  g1[4] = rows;              // tile_dim1 = rows (bits[143:128]); tile_dim2=0
  g1[5] = K;                 // tensor_dim0_stride (bits[207:160], low 32)
  g1[6] = 0;
  g1[7] = 0;
  i32x4 g2 = {0, 0, 0, 0};   // tile_dim3=0 -> unused
  i32x4 g3 = {0, 0, 0, 0};
#if __clang_major__ >= 23
  // amdgpu-toolchain (clang-23 / therock-10.0): 6-arg form, extra int32x8
  // has no ISA D#-group counterpart for a 2D tile -> zeros.
  i32x8 g4 = {0, 0, 0, 0, 0, 0, 0, 0};
  __builtin_amdgcn_tensor_load_to_lds(g0, g1, g2, g3, g4, 0);
#else
  // ROCm 7.2 (clang-22): 5-arg form, groups 0-3 + cpol.
  __builtin_amdgcn_tensor_load_to_lds(g0, g1, g2, g3, 0);
#endif
}
#endif

// ---------------------------------------------------------------------------
// D[M,Nout] = A[M,K] @ W[Nout,K]^T (+ bias), A/W given as hi/lo bf16 pairs.
// Accumulate Ahi*Whi + Ahi*Wlo + Alo*Whi in f32 (3 WMMAs per 16x16x32 step).
// WG tile 64(M) x 128(N), 8 waves in 2x4, each wave 32x32 (2x2 WMMA tiles).
// K staged 32-wide, double-buffered; copies by TDM (wave 0) overlap compute.
// Requires M%64==0, Nout%128==0, K%32==0 (true for all calls here).
// ---------------------------------------------------------------------------
__global__ __launch_bounds__(256) void gemm_bf16x3(
    const unsigned short* __restrict__ Ah,
    const unsigned short* __restrict__ Al,
    const unsigned short* __restrict__ Wh,
    const unsigned short* __restrict__ Wl,
    const float* __restrict__ bias,
    float* __restrict__ D,
    int M, int K, int Nout) {
  __shared__ alignas(16) unsigned short smem[2][TILE_USH];
  // sub-buffer offsets within one stage buffer (bf16 units)
  const int oAh = 0;
  const int oAl = 64 * LDA;
  const int oBh = 128 * LDA;
  const int oBl = 256 * LDA;

  const int tid = threadIdx.x;
  const int m0 = blockIdx.y * 64;
  const int n0 = blockIdx.x * 128;

  const int wave = tid >> 5;
  const int lane = tid & 31;
  const int wm = (wave >> 2) * 32;  // 0,32
  const int wn = (wave & 3) * 32;   // 0..96
  const int lr = lane & 15;
  const int lh = lane >> 4;

  const v8f vzero = {0.f, 0.f, 0.f, 0.f, 0.f, 0.f, 0.f, 0.f};
  v8f acc[2][2] = {{vzero, vzero}, {vzero, vzero}};

  union Frag { v16bf v; int4 q[2]; };

#if USE_TDM
  const unsigned ldsBase = (unsigned)(unsigned long long)&smem[0][0];
  // prologue: fill buffer 0 for k0 = 0
  if (wave == 0) {
    const size_t a0 = (size_t)m0 * K;
    const size_t b0 = (size_t)n0 * K;
    tdm_tile(Ah + a0, ldsBase + oAh * 2, 64, K);
    tdm_tile(Al + a0, ldsBase + oAl * 2, 64, K);
    tdm_tile(Wh + b0, ldsBase + oBh * 2, 128, K);
    tdm_tile(Wl + b0, ldsBase + oBl * 2, 128, K);
    __builtin_amdgcn_s_wait_tensorcnt(0);
  }
  __syncthreads();
#else
  const int srow = tid >> 2;        // 0..63
  const int scol = (tid & 3) * 8;   // 0,8,16,24
#endif

  int buf = 0;
  for (int k0 = 0; k0 < K; k0 += 32) {
#if USE_TDM
    // kick off next stage into the other buffer (safe: it was last read
    // before the barrier that ended the previous iteration)
    if ((k0 + 32 < K) && wave == 0) {
      const unsigned nb = ldsBase + (buf ^ 1) * (TILE_USH * 2);
      const size_t a0 = (size_t)m0 * K + k0 + 32;
      const size_t b0 = (size_t)n0 * K + k0 + 32;
      tdm_tile(Ah + a0, nb + oAh * 2, 64, K);
      tdm_tile(Al + a0, nb + oAl * 2, 64, K);
      tdm_tile(Wh + b0, nb + oBh * 2, 128, K);
      tdm_tile(Wl + b0, nb + oBl * 2, 128, K);
    }
#else
    {
      const size_t ga = (size_t)(m0 + srow) * K + k0 + scol;
      *(int4*)&smem[buf][oAh + srow * LDA + scol] = *(const int4*)&Ah[ga];
      *(int4*)&smem[buf][oAl + srow * LDA + scol] = *(const int4*)&Al[ga];
      const size_t gb0 = (size_t)(n0 + srow) * K + k0 + scol;
      const size_t gb1 = (size_t)(n0 + srow + 64) * K + k0 + scol;
      *(int4*)&smem[buf][oBh + srow * LDA + scol]        = *(const int4*)&Wh[gb0];
      *(int4*)&smem[buf][oBh + (srow + 64) * LDA + scol] = *(const int4*)&Wh[gb1];
      *(int4*)&smem[buf][oBl + srow * LDA + scol]        = *(const int4*)&Wl[gb0];
      *(int4*)&smem[buf][oBl + (srow + 64) * LDA + scol] = *(const int4*)&Wl[gb1];
    }
    __syncthreads();
#endif

    const unsigned short* sb = &smem[buf][0];
    // A fragment (16x32 bf16): lane<16 -> K{0..7,16..23}, lane>=16 -> K{8..15,24..31}
    Frag a_hi[2], a_lo[2], b_hi[2], b_lo[2];
#pragma unroll
    for (int i = 0; i < 2; ++i) {
      const int base = (wm + 16 * i + lr) * LDA + lh * 8;
      a_hi[i].q[0] = *(const int4*)&sb[oAh + base];
      a_hi[i].q[1] = *(const int4*)&sb[oAh + base + 16];
      a_lo[i].q[0] = *(const int4*)&sb[oAl + base];
      a_lo[i].q[1] = *(const int4*)&sb[oAl + base + 16];
    }
    // B fragment (32x16 bf16): col = lane&15, lanes 0-15 hold K{0..7,16..23}, 16-31 K{8..15,24..31}
#pragma unroll
    for (int j = 0; j < 2; ++j) {
      const int base = (wn + 16 * j + lr) * LDA + lh * 8;
      b_hi[j].q[0] = *(const int4*)&sb[oBh + base];
      b_hi[j].q[1] = *(const int4*)&sb[oBh + base + 16];
      b_lo[j].q[0] = *(const int4*)&sb[oBl + base];
      b_lo[j].q[1] = *(const int4*)&sb[oBl + base + 16];
    }

#pragma unroll
    for (int i = 0; i < 2; ++i)
#pragma unroll
      for (int j = 0; j < 2; ++j) {
        acc[i][j] = __builtin_amdgcn_wmma_f32_16x16x32_bf16(
            false, a_lo[i].v, false, b_hi[j].v, (short)0, acc[i][j], false, false);
        acc[i][j] = __builtin_amdgcn_wmma_f32_16x16x32_bf16(
            false, a_hi[i].v, false, b_lo[j].v, (short)0, acc[i][j], false, false);
        acc[i][j] = __builtin_amdgcn_wmma_f32_16x16x32_bf16(
            false, a_hi[i].v, false, b_hi[j].v, (short)0, acc[i][j], false, false);
      }

#if USE_TDM
    if (wave == 0) __builtin_amdgcn_s_wait_tensorcnt(0);
#endif
    __syncthreads();
    buf ^= 1;
  }

  // Epilogue: C/D layout -> VGPR e: lanes0-15 M=e, lanes16-31 M=e+8; N=lane&15
#pragma unroll
  for (int j = 0; j < 2; ++j) {
    const int col = n0 + wn + 16 * j + lr;
    const float bv = bias ? bias[col] : 0.0f;
#pragma unroll
    for (int i = 0; i < 2; ++i) {
      const int mbase = m0 + wm + 16 * i + lh * 8;
#pragma unroll
      for (int e = 0; e < 8; ++e)
        D[(size_t)(mbase + e) * Nout + col] = acc[i][j][e] + bv;
    }
  }
}

// ---------------------------------------------------------------------------
// Per-(b,h) attention: S = q k^T and norms in one streaming pass over N,
// scale S by temp/(||q_d||*||k_e||) (== normalize-then-dot), softmax rows,
// then out = attn @ v with v staged in LDS. Output written as hi/lo bf16
// in [B,N,C] layout, ready for the projection GEMM.
// ---------------------------------------------------------------------------
__global__ __launch_bounds__(256) void attn_kernel(
    const float* __restrict__ qkv,            // [B,N,3C] f32
    const float* __restrict__ temperature,    // [H]
    unsigned short* __restrict__ yh,          // [B,N,C] bf16 hi
    unsigned short* __restrict__ yl) {        // [B,N,C] bf16 lo
  __shared__ float qs[128 * 48];   // q chunk, later reused for v chunk
  __shared__ float ks[128 * 48];
  __shared__ float Sm[48 * 48];
  __shared__ float nrm[96];        // nq[48], nk[48]

  const int tid = threadIdx.x;
  const int b = blockIdx.x >> 4;
  const int h = blockIdx.x & 15;
  const float* base = qkv + (size_t)b * N_ * K3C;

  const int nl = tid >> 1;         // local row 0..127
  const int d0 = (tid & 1) * 24;   // half of the 48 channels

  float accS[9];
  int dIdx[9], eIdx[9];
#pragma unroll
  for (int r = 0; r < 9; ++r) {
    accS[r] = 0.f;
    const int idx = tid + 256 * r;  // 9*256 == 48*48 exactly
    dIdx[r] = idx / 48;
    eIdx[r] = idx % 48;
  }
  float accN = 0.f;

  // ---- Phase 1: Gram matrix + norms, streaming over N in chunks of 128 ----
  for (int c = 0; c < N_; c += 128) {
    const float* qrow = base + (size_t)(c + nl) * K3C + h * DH_ + d0;
#pragma unroll
    for (int d = 0; d < 24; ++d) {
      qs[nl * 48 + d0 + d] = qrow[d];
      ks[nl * 48 + d0 + d] = qrow[d + C_];
    }
    __syncthreads();
#pragma unroll
    for (int r = 0; r < 9; ++r) {
      float s = 0.f;
      for (int n = 0; n < 128; ++n)
        s += qs[n * 48 + dIdx[r]] * ks[n * 48 + eIdx[r]];
      accS[r] += s;
    }
    if (tid < 96) {
      const float* p = (tid < 48) ? qs : ks;
      const int d = (tid < 48) ? tid : tid - 48;
      float s = 0.f;
      for (int n = 0; n < 128; ++n) { const float v = p[n * 48 + d]; s += v * v; }
      accN += s;
    }
    __syncthreads();
  }
#pragma unroll
  for (int r = 0; r < 9; ++r) Sm[tid + 256 * r] = accS[r];  // idx == d*48+e
  if (tid < 96) nrm[tid] = sqrtf(accN);
  __syncthreads();

  // ---- Phase 2: scale + softmax over e (one thread per row d) ----
  if (tid < 48) {
    const float scale_d = temperature[h] / fmaxf(nrm[tid], 1e-12f);
    float row[48];
    float mx = -3.4e38f;
#pragma unroll
    for (int e = 0; e < 48; ++e) {
      const float v = Sm[tid * 48 + e] * scale_d / fmaxf(nrm[48 + e], 1e-12f);
      row[e] = v;
      mx = fmaxf(mx, v);
    }
    float sum = 0.f;
#pragma unroll
    for (int e = 0; e < 48; ++e) { const float v = expf(row[e] - mx); row[e] = v; sum += v; }
    const float inv = 1.0f / sum;
#pragma unroll
    for (int e = 0; e < 48; ++e) Sm[tid * 48 + e] = row[e] * inv;
  }
  __syncthreads();

  // ---- Phase 3: out[d,n] = sum_e attn[d,e] * v[e,n]; write hi/lo bf16 ----
  for (int c = 0; c < N_; c += 128) {
    const float* vrow = base + (size_t)(c + nl) * K3C + 2 * C_ + h * DH_ + d0;
#pragma unroll
    for (int d = 0; d < 24; ++d) qs[nl * 48 + d0 + d] = vrow[d];
    __syncthreads();
    const size_t ob = ((size_t)b * N_ + (c + nl)) * C_ + h * DH_ + d0;
    const float* vv = &qs[nl * 48];
    for (int d = 0; d < 24; ++d) {
      const float* arow = &Sm[(d0 + d) * 48];
      float s = 0.f;
#pragma unroll
      for (int e = 0; e < 48; ++e) s += arow[e] * vv[e];
      const unsigned short hb = f2bf(s);
      yh[ob + d] = hb;
      yl[ob + d] = f2bf(s - bf2f(hb));
    }
    __syncthreads();
  }
}

// ---------------------------------------------------------------------------
extern "C" void kernel_launch(void* const* d_in, const int* in_sizes, int n_in,
                              void* d_out, int out_size, void* d_ws, size_t ws_size,
                              hipStream_t stream) {
  (void)in_sizes; (void)n_in; (void)out_size; (void)ws_size;
  const float* x           = (const float*)d_in[0];  // [B,N,C]
  const float* w_qkv       = (const float*)d_in[1];  // [3C,C]
  const float* w_proj      = (const float*)d_in[2];  // [C,C]
  const float* b_proj      = (const float*)d_in[3];  // [C]
  const float* temperature = (const float*)d_in[4];  // [H,1,1]
  float* out = (float*)d_out;

  const size_t M = (size_t)B_ * N_;      // 32768
  char* ws = (char*)d_ws;
  size_t off = 0;
  auto carve = [&](size_t bytes) -> void* {
    void* p = ws + off;
    off = (off + bytes + 255) & ~(size_t)255;
    return p;
  };
  float*          qkvf = (float*)carve(M * K3C * sizeof(float));       // ~302 MB
  unsigned short* xh   = (unsigned short*)carve(M * C_ * 2);           // 48 MB (reused as y_hi)
  unsigned short* xl   = (unsigned short*)carve(M * C_ * 2);           // 48 MB (reused as y_lo)
  unsigned short* wqh  = (unsigned short*)carve((size_t)K3C * C_ * 2);
  unsigned short* wql  = (unsigned short*)carve((size_t)K3C * C_ * 2);
  unsigned short* wph  = (unsigned short*)carve((size_t)C_ * C_ * 2);
  unsigned short* wpl  = (unsigned short*)carve((size_t)C_ * C_ * 2);

  // 1) split inputs to hi/lo bf16
  split_kernel<<<4096, 256, 0, stream>>>(x, xh, xl, (long)M * C_);
  split_kernel<<<1024, 256, 0, stream>>>(w_qkv, wqh, wql, (long)K3C * C_);
  split_kernel<<<512,  256, 0, stream>>>(w_proj, wph, wpl, (long)C_ * C_);

  // 2) QKV projection: [32768,768] x [768,2304] -> f32 qkv
  gemm_bf16x3<<<dim3(K3C / 128, M / 64), 256, 0, stream>>>(
      xh, xl, wqh, wql, nullptr, qkvf, (int)M, C_, K3C);

  // 3) per-(b,h) XCA core; writes y as hi/lo bf16 into xh/xl (x dead now)
  attn_kernel<<<B_ * H_, 256, 0, stream>>>(qkvf, temperature, xh, xl);

  // 4) output projection with bias: [32768,768] x [768,768] -> d_out
  gemm_bf16x3<<<dim3(C_ / 128, M / 64), 256, 0, stream>>>(
      xh, xl, wph, wpl, b_proj, out, (int)M, C_, C_);
}
